// CrossSparseAggrNet_v2_69595650064758
// MI455X (gfx1250) — matrix operand order repaired and checked by reading, not compiled, not run on hardware
//
#include <hip/hip_runtime.h>
#include <math.h>

// ---------------------------------------------------------------------------
// CrossSparseAggrNet for MI455X (gfx1250, wave32):
//   WMMA f32_16x16x32_f16 GEMMs, K-pair-packed B operands, TDM weight staging.
// ---------------------------------------------------------------------------

typedef __attribute__((ext_vector_type(16))) _Float16 v16h;
typedef __attribute__((ext_vector_type(8)))  float    v8f;
typedef __attribute__((ext_vector_type(4)))  unsigned int u32x4;
typedef __attribute__((ext_vector_type(8)))  int i32x8;
typedef __attribute__((ext_vector_type(4)))  int i32x4;

static constexpr int kBV   = 128;   // images
static constexpr int kBT   = 64;    // captions
static constexpr int kLT   = 60;    // max caption length
static constexpr int kE    = 512;   // embed
static constexpr int kNP   = 196;   // patches
static constexpr int kKEEP = 118;   // ceil(196*0.6)
static constexpr int kTAIL = 78;    // 196-118
static constexpr int kKP   = 58;    // int(196*0.5*0.6)
static constexpr int kHID  = 102;   // int(512*0.2)
static constexpr int kHIDP = 112;   // padded to 7 N-tiles
static constexpr int kMAXKT = 100;
static constexpr float kLNEPS = 1e-5f;

// pair-kernel LDS map (dynamic; exact 320 KB fit = 160 x 2KB blocks)
static constexpr unsigned kOffSel   = 0;        // sel pairs u32 | simgh+selcapp
static constexpr unsigned kOffUnion = 131072;   // LN f16 | logits f32 | sel_img f32
static constexpr unsigned kOffHid   = 262144;   // 128x128 f16 gelu | sim f32
static constexpr unsigned kOffWth   = 294912;   // w2 pairs, then softmax-w f16
static constexpr unsigned kOffPan   = 311296;   // 2 x (16x112 u32) TDM panels
static constexpr unsigned kPanBytes = 16 * kHIDP * 4;  // 7168
static constexpr unsigned kOffMisc  = 325632;   // mu/rsig/red f32
static constexpr size_t   kPairLds  = 327680;

// ------------------------------- helpers -----------------------------------

__device__ __forceinline__ float wave_sum(float v) {
#pragma unroll
  for (int o = 16; o > 0; o >>= 1) v += __shfl_down(v, o, 32);
  return v;  // lane 0 holds full sum
}

__device__ __forceinline__ float gelu_exact(float x) {
  return 0.5f * x * (1.0f + erff(x * 0.70710678118654752f));
}

__device__ __forceinline__ unsigned pack_h2(float a, float b) {
  union { unsigned u; _Float16 h[2]; } r;
  r.h[0] = (_Float16)a;
  r.h[1] = (_Float16)b;
  return r.u;
}

__device__ __forceinline__ v8f wmma16(v16h a, v16h b, v8f c) {
  return __builtin_amdgcn_wmma_f32_16x16x32_f16(
      /*neg_a=*/false, a, /*neg_b=*/false, b,
      /*c_mod=*/(short)0, c, /*reuse_a=*/false, /*reuse_b=*/false);
}

// A fragment (16x32 f16) from row-major f16 LDS; k = 16*(lane/16) + s order.
// K-pairs are contiguous -> compiler merges into ds_load_b128.
__device__ __forceinline__ v16h load_a_lds(const _Float16* base, int row0,
                                           int k0, int ld) {
  int l = threadIdx.x & 31;
  int m = row0 + (l & 15);
  int kh = k0 + ((l >> 4) << 4);
  v16h a;
#pragma unroll
  for (int v = 0; v < 8; ++v) {
    int kk = kh + (v << 1);
    a[2 * v]     = base[m * ld + kk];
    a[2 * v + 1] = base[m * ld + kk + 1];
  }
  return a;
}

// B fragment (32x16 f16) from K-pair-packed LDS:
//   P[(k/2)*ldn + n] = { f16 B(k,n), f16 B(k+1,n) }   (k even)
// -> 8x ds_load_b32, conflict-free (lanes walk consecutive dwords).
__device__ __forceinline__ v16h load_bp_lds(const unsigned* P, int k0,
                                            int col0, int ldn) {
  int l = threadIdx.x & 31;
  int n = col0 + (l & 15);
  int ph = (k0 >> 1) + ((l >> 4) << 3);  // pair-row base (+8 for lanes 16-31)
  union { unsigned u[8]; v16h h; } r;
#pragma unroll
  for (int v = 0; v < 8; ++v) r.u[v] = P[(ph + v) * ldn + n];
  return r.h;
}

// --- Tensor Data Mover: 2-D tile load, global -> LDS (D# per ISA sec 8) -----
__device__ __forceinline__ void tdm_load_2d(unsigned lds_off, const void* gaddr,
                                            unsigned dsize_code,  // 1=2B,2=4B
                                            unsigned tile_d0,     // elems
                                            unsigned tile_d1,     // rows
                                            unsigned tensor_d0,
                                            unsigned tensor_d1,
                                            unsigned stride0) {
  unsigned long long ga = (unsigned long long)(uintptr_t)gaddr;
  u32x4 g0;
  g0[0] = 1u;                                            // count=1 (user D#)
  g0[1] = lds_off;                                       // lds_addr (bytes)
  g0[2] = (unsigned)(ga & 0xFFFFFFFFu);                  // global_addr[31:0]
  g0[3] = (unsigned)((ga >> 32) & 0x1FFFFFFu) | (2u << 30);  // [56:32]|type=2
  i32x8 g1;
  g1[0] = (int)(dsize_code << 16);                       // data_size
  g1[1] = (int)((tensor_d0 & 0xFFFFu) << 16);            // dim0[15:0]
  g1[2] = (int)(((tensor_d0 >> 16) & 0xFFFFu) | ((tensor_d1 & 0xFFFFu) << 16));
  g1[3] = (int)(((tensor_d1 >> 16) & 0xFFFFu) | ((tile_d0 & 0xFFFFu) << 16));
  g1[4] = (int)(tile_d1 & 0xFFFFu);                      // tile_dim1, dim2=0
  g1[5] = (int)stride0;                                  // dim0_stride[31:0]
  g1[6] = 0;
  g1[7] = 0;
  i32x4 z4 = {0, 0, 0, 0};
#if __clang_major__ >= 23
  i32x8 z8 = {0, 0, 0, 0, 0, 0, 0, 0};
  __builtin_amdgcn_tensor_load_to_lds(g0, g1, z4, z4, z8, 0);
#else
  __builtin_amdgcn_tensor_load_to_lds(g0, g1, z4, z4, 0);
#endif
}

// ---------------------------------------------------------------------------
// Kernel 0: pair-pack + pad weights.
//   w1p: [256 pair-rows][112] u32, pair p = rows 2p,2p+1 of ta_w1 (512x102)
//   w2p: [64  pair-rows][ 64] u32, from ta_w2 (102x58)
// ---------------------------------------------------------------------------
__global__ __launch_bounds__(256) void prep_weights_kernel(
    const float* __restrict__ w1, const float* __restrict__ w2,
    unsigned* __restrict__ w1p, unsigned* __restrict__ w2p) {
  int idx = blockIdx.x * 256 + threadIdx.x;
  int stride = gridDim.x * 256;
  for (int i = idx; i < 256 * kHIDP; i += stride) {
    int p = i / kHIDP, n = i % kHIDP;
    float f0 = (n < kHID) ? w1[(2 * p) * kHID + n] : 0.f;
    float f1 = (n < kHID) ? w1[(2 * p + 1) * kHID + n] : 0.f;
    w1p[i] = pack_h2(f0, f1);
  }
  for (int i = idx; i < 64 * 64; i += stride) {
    int p = i / 64, n = i % 64;
    int k0 = 2 * p, k1 = 2 * p + 1;
    float f0 = (k0 < kHID && n < kKP) ? w2[k0 * kKP + n] : 0.f;
    float f1 = (k1 < kHID && n < kKP) ? w2[k1 * kKP + n] : 0.f;
    w2p[i] = pack_h2(f0, f1);
  }
}

// ---------------------------------------------------------------------------
// Kernel 1: per-image patch inv-norms, normalized global vec, self_attn
// ---------------------------------------------------------------------------
__global__ __launch_bounds__(256) void img_stats_kernel(
    const float* __restrict__ img, float* __restrict__ invn_ws,
    float* __restrict__ sattn_ws) {
  __shared__ float glo[kE];
  __shared__ float red[8];
  int b = blockIdx.x, tid = threadIdx.x;
  int wave = tid >> 5, lane = tid & 31;
  const float* sp = img + ((size_t)b * (kNP + 1) + 1) * kE;

  for (int c = tid; c < kE; c += 256) {
    float s = 0.f;
    for (int p = 0; p < kNP; ++p) s += sp[(size_t)p * kE + c];
    glo[c] = s * (1.0f / kNP);
  }
  __syncthreads();
  float ss = 0.f;
  for (int c = tid; c < kE; c += 256) { float v = glo[c]; ss += v * v; }
  ss = wave_sum(ss);
  if (lane == 0) red[wave] = ss;
  __syncthreads();
  if (tid == 0) {
    float t = 0.f;
    for (int w = 0; w < 8; ++w) t += red[w];
    red[0] = 1.0f / fmaxf(sqrtf(t), 1e-12f);
  }
  __syncthreads();
  float gscale = red[0];

  for (int p = wave; p < kNP; p += 8) {
    float s2 = 0.f, dp = 0.f;
    const float* row = sp + (size_t)p * kE;
    for (int c = lane; c < kE; c += 32) {
      float x = row[c];
      s2 += x * x;
      dp += x * glo[c];
    }
    s2 = wave_sum(s2);
    dp = wave_sum(dp);
    if (lane == 0) {
      float inv = 1.0f / fmaxf(sqrtf(s2), 1e-12f);
      invn_ws[b * kNP + p] = inv;
      sattn_ws[b * kNP + p] = dp * gscale * inv;
    }
  }
}

// ---------------------------------------------------------------------------
// Kernel 2: per-caption global vec + caption aggregation -> sel_cap
// ---------------------------------------------------------------------------
static constexpr size_t kCapLds =
    (size_t)(59 * kE + 59 * kHID + 59 * kMAXKT + kMAXKT * 64 + kE + 16) * 4;

__global__ __launch_bounds__(256) void cap_stats_kernel(
    const float* __restrict__ cap, const int* __restrict__ lens,
    const float* __restrict__ lng, const float* __restrict__ lnb,
    const float* __restrict__ w1, const float* __restrict__ b1,
    const float* __restrict__ fc2w, const float* __restrict__ fc2b,
    const float* __restrict__ cscale_p, float* __restrict__ capglo_ws,
    float* __restrict__ selcap_ws) {
  extern __shared__ char smem[];
  float* lnc     = (float*)smem;              // 59*512
  float* hidL    = lnc + 59 * kE;             // 59*102
  float* logitsL = hidL + 59 * kHID;          // 59*100
  float* cwL     = logitsL + 59 * kMAXKT;     // 100*64
  float* capg    = cwL + kMAXKT * 64;         // 512
  float* red     = capg + kE;                 // 16

  int i = blockIdx.x, tid = threadIdx.x;
  int wave = tid >> 5, lane = tid & 31;
  int n = lens[i];
  int m = n - 1;
  int kt = m / 2; if (kt < 4) kt = 4; if (kt > kMAXKT) kt = kMAXKT;
  const float* ci = cap + (size_t)i * kLT * kE;

  for (int c = tid; c < kE; c += 256) {
    float s = 0.f;
    for (int t = 0; t < n; ++t) s += ci[(size_t)t * kE + c];
    capg[c] = s / (float)n;
  }
  __syncthreads();
  float ss = 0.f;
  for (int c = tid; c < kE; c += 256) { float v = capg[c]; ss += v * v; }
  ss = wave_sum(ss);
  if (lane == 0) red[wave] = ss;
  __syncthreads();
  if (tid == 0) {
    float t = 0.f;
    for (int w = 0; w < 8; ++w) t += red[w];
    red[0] = 1.0f / fmaxf(sqrtf(t), 1e-12f);
  }
  __syncthreads();
  float gs = red[0];
  for (int c = tid; c < kE; c += 256) capglo_ws[i * kE + c] = capg[c] * gs;

  for (int r = wave; r < m; r += 8) {
    const float* row = ci + (size_t)(1 + r) * kE;
    float s = 0.f, s2 = 0.f;
    for (int c = lane; c < kE; c += 32) { float x = row[c]; s += x; s2 += x * x; }
    s = wave_sum(s); s2 = wave_sum(s2);
    s = __shfl(s, 0, 32); s2 = __shfl(s2, 0, 32);
    float mu = s / (float)kE;
    float rs = rsqrtf(s2 / (float)kE - mu * mu + kLNEPS);
    for (int c = lane; c < kE; c += 32)
      lnc[r * kE + c] = (row[c] - mu) * rs * lng[c] + lnb[c];
  }
  __syncthreads();

  for (int o = tid; o < m * kHID; o += 256) {
    int r = o / kHID, h = o % kHID;
    float acc = b1[h];
    for (int c = 0; c < kE; ++c) acc += lnc[r * kE + c] * w1[c * kHID + h];
    hidL[o] = gelu_exact(acc);
  }
  __syncthreads();

  for (int o = tid; o < m * kt; o += 256) {
    int r = o / kt, j = o % kt;
    float acc = fc2b[i * kMAXKT + j];
    for (int h = 0; h < kHID; ++h)
      acc += hidL[r * kHID + h] * fc2w[((size_t)i * kHID + h) * kMAXKT + j];
    logitsL[r * kMAXKT + j] = acc;
  }
  __syncthreads();

  float scl = cscale_p[0];
  for (int j = tid; j < kt; j += 256) {
    float mx = -1e30f;
    for (int r = 0; r < m; ++r) mx = fmaxf(mx, logitsL[r * kMAXKT + j] * scl);
    float den = 0.f;
    for (int r = 0; r < m; ++r) den += expf(logitsL[r * kMAXKT + j] * scl - mx);
    float rd = 1.f / den;
    for (int r = 0; r < m; ++r)
      cwL[j * 64 + r] = expf(logitsL[r * kMAXKT + j] * scl - mx) * rd;
  }
  __syncthreads();

  float* out = selcap_ws + (size_t)i * (kMAXKT + 1) * kE;
  for (int r = wave; r < kt + 1; r += 8) {
    float vals[16];
#pragma unroll
    for (int q = 0; q < 16; ++q) {
      int c = lane + q * 32;
      if (r == 0) {
        vals[q] = ci[c];
      } else {
        float a = 0.f;
        for (int t = 0; t < m; ++t)
          a += cwL[(r - 1) * 64 + t] * ci[(size_t)(1 + t) * kE + c];
        vals[q] = a;
      }
    }
    float s2 = 0.f;
#pragma unroll
    for (int q = 0; q < 16; ++q) s2 += vals[q] * vals[q];
    s2 = wave_sum(s2); s2 = __shfl(s2, 0, 32);
    float inv = 1.f / fmaxf(sqrtf(s2), 1e-12f);
#pragma unroll
    for (int q = 0; q < 16; ++q) out[(size_t)r * kE + lane + q * 32] = vals[q] * inv;
  }
}

// ---------------------------------------------------------------------------
// Kernel 3: per pair -- scores, stable rank-sort, keep idx, tail softmax, extra
// ---------------------------------------------------------------------------
__global__ __launch_bounds__(256) void score_sort_kernel(
    const float* __restrict__ img, const float* __restrict__ capglo_ws,
    const float* __restrict__ invn_ws, const float* __restrict__ sattn_ws,
    int* __restrict__ keep_ws, float* __restrict__ extra_ws) {
  __shared__ float score[kNP];
  __shared__ int order[kNP];
  __shared__ float cg[kE];
  __shared__ float nw[kTAIL];
  __shared__ float mred[2];
  int b = blockIdx.x, i = blockIdx.y, tid = threadIdx.x;
  int wave = tid >> 5, lane = tid & 31;
  const float* sp = img + ((size_t)b * (kNP + 1) + 1) * kE;
  size_t pair = (size_t)i * kBV + b;

  for (int c = tid; c < kE; c += 256) cg[c] = capglo_ws[i * kE + c];
  __syncthreads();

  for (int p = wave; p < kNP; p += 8) {
    float dp = 0.f;
    const float* row = sp + (size_t)p * kE;
    for (int c = lane; c < kE; c += 32) dp += row[c] * cg[c];
    dp = wave_sum(dp);
    if (lane == 0)
      score[p] = sattn_ws[b * kNP + p] + dp * invn_ws[b * kNP + p];
  }
  __syncthreads();

  if (tid < kNP) {  // stable descending rank == jnp.argsort(-score)
    float s = score[tid];
    int r = 0;
    for (int q = 0; q < kNP; ++q) {
      float sq = score[q];
      r += (sq > s) || (sq == s && q < tid);
    }
    order[r] = tid;
  }
  __syncthreads();

  if (tid < kKEEP) keep_ws[pair * kKEEP + tid] = order[tid];
  if (tid == 0) {
    float mx = -1e30f;
    for (int j = kKEEP; j < kNP; ++j) mx = fmaxf(mx, score[order[j]]);
    float den = 0.f;
    for (int j = kKEEP; j < kNP; ++j) den += expf(score[order[j]] - mx);
    mred[0] = mx;
    mred[1] = 1.f / den;
  }
  __syncthreads();
  if (tid < kTAIL)
    nw[tid] = expf(score[order[kKEEP + tid]] - mred[0]) * mred[1];
  __syncthreads();

  for (int c = tid; c < kE; c += 256) {
    float e = 0.f;
    for (int j = 0; j < kTAIL; ++j)
      e += nw[j] * sp[(size_t)order[kKEEP + j] * kE + c];
    extra_ws[pair * kE + c] = e;
  }
}

// ---------------------------------------------------------------------------
// Kernel 4: fused per-pair WMMA pipeline, TDM staging, pair-packed B operands
// ---------------------------------------------------------------------------
__global__ __launch_bounds__(256) void pair_sim_kernel(
    const float* __restrict__ img, const int* __restrict__ lens,
    const float* __restrict__ lng, const float* __restrict__ lnb,
    const unsigned* __restrict__ w1p, const float* __restrict__ b1,
    const unsigned* __restrict__ w2p, const float* __restrict__ b2,
    const float* __restrict__ tscale_p, const int* __restrict__ keep_ws,
    const float* __restrict__ extra_ws, const float* __restrict__ selcap_ws,
    float* __restrict__ out) {
  extern __shared__ char smem[];
  unsigned* sel2    = (unsigned*)(smem + kOffSel);   // [64 pair-rows][512] u32
  _Float16* simgh   = (_Float16*)(smem + kOffSel);   // phase2: [64][512] f16
  unsigned* selcapp = (unsigned*)(smem + kOffSel + 65536);  // [256][32] u32
  _Float16* lnh  = (_Float16*)(smem + kOffUnion);    // 128x512 f16
  float* logits  = (float*)(smem + kOffUnion);       // 128x64 f32
  float* simg    = (float*)(smem + kOffUnion);       // 60x512 f32
  _Float16* hidh = (_Float16*)(smem + kOffHid);      // 128x128 f16
  float* simbuf  = (float*)(smem + kOffHid);         // [64][32] f32
  _Float16* wth  = (_Float16*)(smem + kOffWth);      // softmax-w f16
  unsigned* w2l  = (unsigned*)(smem + kOffWth);      // phase1: w2 pairs
  float* mu   = (float*)(smem + kOffMisc);           // 128
  float* rsig = mu + 128;                            // 128
  float* redf = rsig + 128;                          // 8

  int b = blockIdx.x, i = blockIdx.y;
  int tid = threadIdx.x, wave = tid >> 5, lane = tid & 31;
  size_t pair = (size_t)i * kBV + b;
  const int* keep = keep_ws + pair * kKEEP;
  const float* sp = img + ((size_t)b * (kNP + 1) + 1) * kE;

  // TDM: whole pair-packed ta_w2 into wth region + first ta_w1 K-panel,
  // overlapped with the gather/LN stage.
  if (wave == 0) {
    tdm_load_2d(kOffWth, w2p, 2, 64, 64, 64, 64, 64);
    tdm_load_2d(kOffPan, w1p, 2, kHIDP, 16, kHIDP, 256, kHIDP);
  }
  __builtin_prefetch(lng + lane * 16, 0, 3);

  // ---- stage A: gather sel row-pairs (f32 -> packed f16x2) + LN stats ------
  for (int rp = wave; rp < 64; rp += 8) {
    int r0 = 2 * rp, r1 = r0 + 1;
    bool v0 = r0 < kKEEP, v1 = r1 < kKEEP;
    const float* row0 = v0 ? sp + (size_t)keep[r0] * kE : sp;
    const float* row1 = v1 ? sp + (size_t)keep[r1] * kE : sp;
    float s0 = 0.f, q0 = 0.f, s1 = 0.f, q1 = 0.f;
#pragma unroll
    for (int q = 0; q < 16; ++q) {
      int c = lane + q * 32;
      float x0 = v0 ? row0[c] : 0.f;
      float x1 = v1 ? row1[c] : 0.f;
      sel2[rp * kE + c] = pack_h2(x0, x1);
      s0 += x0; q0 += x0 * x0;
      s1 += x1; q1 += x1 * x1;
    }
    s0 = wave_sum(s0); q0 = wave_sum(q0);
    s1 = wave_sum(s1); q1 = wave_sum(q1);
    if (lane == 0) {
      float m0 = s0 / (float)kE, m1 = s1 / (float)kE;
      mu[r0] = m0;
      mu[r1] = m1;
      rsig[r0] = v0 ? rsqrtf(q0 / (float)kE - m0 * m0 + kLNEPS) : 0.f;
      rsig[r1] = v1 ? rsqrtf(q1 / (float)kE - m1 * m1 + kLNEPS) : 0.f;
    }
  }
  __syncthreads();

  // ---- LN into lnh (f16); zero hid pad -------------------------------------
  for (int r = wave; r < 128; r += 8) {
    float mm = mu[r], rs = rsig[r];
#pragma unroll
    for (int q = 0; q < 16; ++q) {
      int c = lane + q * 32;
      union { unsigned u; _Float16 h[2]; } pv;
      pv.u = sel2[(r >> 1) * kE + c];
      float x = (float)pv.h[r & 1];
      float y = (r < kKEEP) ? ((x - mm) * rs * lng[c] + lnb[c]) : 0.f;
      lnh[r * kE + c] = (_Float16)y;
    }
  }
  for (int idx = tid; idx < 128 * 128; idx += 256) hidh[idx] = (_Float16)0.f;
  __syncthreads();

  // ---- G1: hid = gelu(LN(sel) @ ta_w1 + b1)   M=128 N=112 K=512 ------------
  v8f acc1[7];
  v8f zero8 = {};
#pragma unroll
  for (int j = 0; j < 7; ++j) acc1[j] = zero8;
  for (int kp = 0; kp < 16; ++kp) {
    if (wave == 0) {
      if (kp + 1 < 16) {
        tdm_load_2d(kOffPan + ((kp + 1) & 1) * kPanBytes,
                    w1p + (size_t)(kp + 1) * 16 * kHIDP, 2, kHIDP, 16, kHIDP,
                    256, kHIDP);
        __builtin_amdgcn_s_wait_tensorcnt(1);  // panel kp (and w2p) landed
      } else {
        __builtin_amdgcn_s_wait_tensorcnt(0);
      }
    }
    __syncthreads();
    const unsigned* pan = (const unsigned*)(smem + kOffPan + (kp & 1) * kPanBytes);
#pragma unroll
    for (int j = 0; j < 7; ++j) {
      int t = j * 8 + wave;
      int mt = t / 7, nt = t % 7;
      v16h a = load_a_lds(lnh, mt * 16, kp * 32, kE);
      v16h bb = load_bp_lds(pan, 0, nt * 16, kHIDP);
      acc1[j] = wmma16(a, bb, acc1[j]);
    }
    __syncthreads();
  }
#pragma unroll
  for (int j = 0; j < 7; ++j) {
    int t = j * 8 + wave;
    int mt = t / 7, nt = t % 7;
    int n = nt * 16 + (lane & 15);
    int rb = mt * 16 + ((lane >> 4) << 3);
    float bias = (n < kHID) ? b1[n] : 0.f;
#pragma unroll
    for (int v = 0; v < 8; ++v) {
      int r = rb + v;
      float x = acc1[j][v] + bias;
      hidh[r * 128 + n] =
          (_Float16)((r < kKEEP && n < kHID) ? gelu_exact(x) : 0.f);
    }
  }
  __syncthreads();

  // ---- G2: logits = hid @ ta_w2 + b2   M=128 N=64 K=128 (B pairs via TDM) --
  for (int t = wave; t < 8 * 4; t += 8) {
    int mt = t >> 2, nt = t & 3;
    v8f acc = zero8;
    for (int ks = 0; ks < 128; ks += 32) {
      v16h a = load_a_lds(hidh, mt * 16, ks, 128);
      v16h bb = load_bp_lds(w2l, ks, nt * 16, 64);
      acc = wmma16(a, bb, acc);
    }
    int n = nt * 16 + (lane & 15);
    int rb = mt * 16 + ((lane >> 4) << 3);
    float bias = (n < kKP) ? b2[n] : 0.f;
#pragma unroll
    for (int v = 0; v < 8; ++v) logits[(rb + v) * 64 + n] = acc[v] + bias;
  }
  __syncthreads();

  // ---- transposed softmax: wth[p][n] over n (118 kept patches) -------------
  for (int idx = tid; idx < 64 * 128; idx += 256) wth[idx] = (_Float16)0.f;
  __syncthreads();
  float tscale = tscale_p[0];
  for (int p = tid; p < kKP; p += 256) {
    float mx = -1e30f;
    for (int nn = 0; nn < kKEEP; ++nn)
      mx = fmaxf(mx, logits[nn * 64 + p] * tscale);
    float den = 0.f;
    for (int nn = 0; nn < kKEEP; ++nn)
      den += expf(logits[nn * 64 + p] * tscale - mx);
    float rd = 1.f / den;
    for (int nn = 0; nn < kKEEP; ++nn)
      wth[p * 128 + nn] =
          (_Float16)(expf(logits[nn * 64 + p] * tscale - mx) * rd);
  }
  __syncthreads();

  // ---- sel_img rows 0 (cls) / 59 (extra); G3: aggr = w @ sel ---------------
  for (int c = tid; c < kE; c += 256) {
    simg[0 * kE + c] = img[(size_t)b * (kNP + 1) * kE + c];
    simg[59 * kE + c] = extra_ws[pair * kE + c];
  }
  for (int t = wave; t < 4 * 32; t += 8) {
    int mt = t >> 5, nt = t & 31;
    v8f acc = zero8;
    for (int ks = 0; ks < 128; ks += 32) {
      v16h a = load_a_lds(wth, mt * 16, ks, 128);
      v16h bb = load_bp_lds(sel2, ks, nt * 16, kE);
      acc = wmma16(a, bb, acc);
    }
    int n = nt * 16 + (lane & 15);
    int rb = mt * 16 + ((lane >> 4) << 3);
#pragma unroll
    for (int v = 0; v < 8; ++v) {
      int r = rb + v;
      if (r < kKP) simg[(1 + r) * kE + n] = acc[v];
    }
  }
  __syncthreads();

  // ---- l2norm the 60 sel_img rows ------------------------------------------
  for (int r = wave; r < 60; r += 8) {
    float s2 = 0.f;
#pragma unroll
    for (int q = 0; q < 16; ++q) {
      float x = simg[r * kE + lane + q * 32];
      s2 += x * x;
    }
    s2 = wave_sum(s2); s2 = __shfl(s2, 0, 32);
    float inv = 1.f / fmaxf(sqrtf(s2), 1e-12f);
#pragma unroll
    for (int q = 0; q < 16; ++q) simg[r * kE + lane + q * 32] *= inv;
  }
  __syncthreads();

  // ---- stage G4 operands (sel region free after G3):
  //   A = simgh [64][512] f16 (rows 60..63 zero)
  //   B = selcapp pair-packed sel_cap [256][32] u32
  int nword = lens[i];
  int mw = nword - 1;
  int kt = mw / 2; if (kt < 4) kt = 4; if (kt > kMAXKT) kt = kMAXKT;
  int rows = kt + 1;
  if (rows > 32) rows = 32;  // actual rows <= 30 for L_T=60
  int mtiles = (rows + 15) >> 4;
  const float* scap = selcap_ws + (size_t)i * (kMAXKT + 1) * kE;
  for (int idx = tid; idx < 64 * 512; idx += 256) {
    int n = idx >> 9, k = idx & 511;
    simgh[idx] = (_Float16)((n < 60) ? simg[n * kE + k] : 0.f);
  }
  __syncthreads();  // simgh overwrites sel2; also simg must be fully read
  for (int idx = tid; idx < 256 * 32; idx += 256) {
    int p = idx >> 5, t = idx & 31;
    unsigned val = 0u;
    if (t < rows) {
      int k0 = 2 * p;
      val = pack_h2(scap[(size_t)t * kE + k0], scap[(size_t)t * kE + k0 + 1]);
    }
    selcapp[idx] = val;
  }
  __syncthreads();

  // ---- G4 (transposed): sim^T = sel_img @ sel_cap^T  M=64 N=rows K=512 -----
  for (int t = wave; t < 4 * 2; t += 8) {
    int mt = t >> 1, nt = t & 1;
    if (nt < mtiles) {
      v8f acc = zero8;
      for (int ks = 0; ks < kE; ks += 32) {
        v16h a = load_a_lds(simgh, mt * 16, ks, kE);
        v16h bb = load_bp_lds(selcapp, ks, nt * 16, 32);
        acc = wmma16(a, bb, acc);
      }
      int n = nt * 16 + (lane & 15);
      int rb = mt * 16 + ((lane >> 4) << 3);
#pragma unroll
      for (int v = 0; v < 8; ++v) simbuf[(rb + v) * 32 + n] = acc[v];
    }
  }
  __syncthreads();

  // ---- mean over caption tokens of max over 60 image tokens ----------------
  float pm = 0.f;
  if (tid < rows) {
    float mx = -1e30f;
    for (int n = 0; n < 60; ++n) mx = fmaxf(mx, simbuf[n * 32 + tid]);
    pm = mx;
  }
  pm = wave_sum(pm);
  if (lane == 0) redf[wave] = pm;
  __syncthreads();
  if (tid == 0) {
    float s = 0.f;
    for (int w = 0; w < 8; ++w) s += redf[w];
    out[(size_t)b * kBT + i] = s / (float)rows;
  }
}

// ---------------------------------------------------------------------------
// Host launch
// ---------------------------------------------------------------------------
extern "C" void kernel_launch(void* const* d_in, const int* in_sizes, int n_in,
                              void* d_out, int out_size, void* d_ws,
                              size_t ws_size, hipStream_t stream) {
  (void)in_sizes; (void)n_in; (void)out_size; (void)ws_size;
  const float* img      = (const float*)d_in[0];
  const float* cap      = (const float*)d_in[1];
  const int*   lens     = (const int*)d_in[2];
  const float* ta_ln_g  = (const float*)d_in[3];
  const float* ta_ln_b  = (const float*)d_in[4];
  const float* ta_w1    = (const float*)d_in[5];
  const float* ta_b1    = (const float*)d_in[6];
  const float* ta_w2    = (const float*)d_in[7];
  const float* ta_b2    = (const float*)d_in[8];
  const float* ta_scale = (const float*)d_in[9];
  const float* ca_ln_g  = (const float*)d_in[10];
  const float* ca_ln_b  = (const float*)d_in[11];
  const float* ca_w1    = (const float*)d_in[12];
  const float* ca_b1    = (const float*)d_in[13];
  const float* ca_scale = (const float*)d_in[14];
  const float* fc2_w    = (const float*)d_in[15];
  const float* fc2_b    = (const float*)d_in[16];
  float* out = (float*)d_out;

  // workspace carve (~34 MB)
  float* ws = (float*)d_ws;
  size_t o = 0;
  float* invn   = ws + o; o += (size_t)kBV * kNP;
  float* sattn  = ws + o; o += (size_t)kBV * kNP;
  float* capglo = ws + o; o += (size_t)kBT * kE;
  float* selcap = ws + o; o += (size_t)kBT * (kMAXKT + 1) * kE;
  int*   keepw  = (int*)(ws + o); o += (size_t)kBT * kBV * kKEEP;
  float* extraw = ws + o; o += (size_t)kBT * kBV * kE;
  unsigned* w1p = (unsigned*)(ws + o);          // 256*112 u32 (pair-packed)
  unsigned* w2p = w1p + (size_t)256 * kHIDP;    // 64*64   u32

  prep_weights_kernel<<<64, 256, 0, stream>>>(ta_w1, ta_w2, w1p, w2p);
  img_stats_kernel<<<kBV, 256, 0, stream>>>(img, invn, sattn);
  cap_stats_kernel<<<kBT, 256, kCapLds, stream>>>(
      cap, lens, ca_ln_g, ca_ln_b, ca_w1, ca_b1, fc2_w, fc2_b, ca_scale,
      capglo, selcap);
  score_sort_kernel<<<dim3(kBV, kBT), 256, 0, stream>>>(
      img, capglo, invn, sattn, keepw, extraw);
  pair_sim_kernel<<<dim3(kBV, kBT), 256, kPairLds, stream>>>(
      img, lens, ta_ln_g, ta_ln_b, w1p, ta_b1, w2p, ta_b2, ta_scale, keepw,
      extraw, selcap, out);
}
